// Sep_GRU_GNN_11433202942554
// MI455X (gfx1250) — compile-verified, matrix operand order checked by reading, not run on hardware
//
#include <hip/hip_runtime.h>

#define NN 100000
#define EE 1600000
#define SS 16
#define DD 32
#define TT 16
#define HH 64

typedef __attribute__((ext_vector_type(16))) __bf16 v16bf;
typedef __attribute__((ext_vector_type(2)))  __bf16 v2bf;
typedef __attribute__((ext_vector_type(8)))  float  v8f;

union FragU { unsigned u[8]; v16bf v; };

// Pack two f32 -> bf16x2: bias-round + one v_perm_b32 (3 VALU ops).
__device__ __forceinline__ unsigned pk_bf16(float a, float b) {
#if __has_builtin(__builtin_amdgcn_cvt_pk_bf16_f32)
  union { v2bf v; unsigned u; } cv;
  cv.v = __builtin_amdgcn_cvt_pk_bf16_f32(a, b);
  return cv.u;
#else
  const unsigned ua = __float_as_uint(a) + 0x8000u;
  const unsigned ub = __float_as_uint(b) + 0x8000u;
  // {b.hi16, a.hi16}: bytes 2,3 of src1(=ua) -> low half; bytes 2,3 of src0(=ub) -> high half
  return __builtin_amdgcn_perm(ub, ua, 0x07060302u);
#endif
}

__device__ __forceinline__ v16bf load_frag_lds(const unsigned* __restrict__ p) {
  FragU f;
#pragma unroll
  for (int i = 0; i < 8; ++i) f.u[i] = p[i];
  return f.v;
}

// Hardware tanh if available; otherwise v_exp + v_rcp form (~5 VALU).
__device__ __forceinline__ float fast_tanh(float x) {
#if __has_builtin(__builtin_amdgcn_tanhf)
  return __builtin_amdgcn_tanhf(x);
#else
  const float e = __expf(2.f * x);
  return 1.f - 2.f * __builtin_amdgcn_rcpf(e + 1.f);
#endif
}
__device__ __forceinline__ float fast_sig(float x) {
  return __builtin_amdgcn_rcpf(1.f + __expf(-x));
}

// K-pair base for 16-bit A/B fragments (lane-half fh, dword v)
__device__ __forceinline__ int kpair(int v, int fh) {
  return ((v & 4) << 2) | ((v & 3) << 1) | (fh << 3);
}

#define WMMA_BF16(A, B, C) \
  __builtin_amdgcn_wmma_f32_16x16x32_bf16(false, (A), false, (B), (short)0, (C), false, false)

// --------------------------------------------------------------------------
// GRU kernel: one wave = 16 nodes (WMMA M-tile), 4 waves per block.
// Per step, per column-tile jj (16 gate columns): r/z get 3 chained WMMAs
// (h-lo, h-hi, x), xn gets 1, hn gets 2 -> 9 WMMAs x 4 jj = 36 per step.
// Only 4 accumulators live at once (32 VGPRs) -> stays in the 256-VGPR window.
// --------------------------------------------------------------------------
#define GRU_WAVES 4
__global__ __launch_bounds__(32 * GRU_WAVES)
void gru_kernel(const float* __restrict__ x,
                const float* __restrict__ Wih, const float* __restrict__ Whh,
                const float* __restrict__ bih, const float* __restrict__ bhh,
                float* __restrict__ gruh)
{
  __shared__ unsigned s_bfrag[36 * 256];        // 36 B-tiles, 32 lanes x 8 dwords
  __shared__ float    s_h[GRU_WAVES][16 * HH];  // per-wave h (C-layout <-> A-layout bridge)

  const int tid   = threadIdx.x;
  const int lane  = tid & 31;
  const int wave  = tid >> 5;
  const int c     = lane & 15;   // C-fragment column
  const int halfc = lane >> 4;   // C-fragment row-half

  for (int i = lane; i < 16 * HH; i += 32) s_h[wave][i] = 0.f;

  // Tiles 0..11: Wih g-tiles (K=d 0..31). Tiles 12+2*j+kh: Whh g-tile j, K-half kh.
  for (int idx = tid; idx < 36 * 32; idx += blockDim.x) {
    const int fl = idx & 31;
    const int tile = idx >> 5;
    const int jcol = fl & 15;
    const int fh = fl >> 4;
    unsigned frag[8];
#pragma unroll
    for (int v = 0; v < 8; ++v) {
      const int k0 = kpair(v, fh);
      float f0, f1;
      if (tile < 12) {
        const int g = tile * 16 + jcol;
        f0 = Wih[g * DD + k0];
        f1 = Wih[g * DD + k0 + 1];
      } else {
        const int tt2 = tile - 12;
        const int j = tt2 >> 1, kh = tt2 & 1;
        const int g = j * 16 + jcol;
        f0 = Whh[g * HH + kh * 32 + k0];
        f1 = Whh[g * HH + kh * 32 + k0 + 1];
      }
      frag[v] = pk_bf16(f0, f1);
    }
#pragma unroll
    for (int v = 0; v < 8; ++v) s_bfrag[tile * 256 + fl * 8 + v] = frag[v];
  }
  __syncthreads();

  float br[4], bz[4], bxn[4], bhn[4];
#pragma unroll
  for (int jj = 0; jj < 4; ++jj) {
    br[jj]  = bih[jj * 16 + c]      + bhh[jj * 16 + c];
    bz[jj]  = bih[64 + jj * 16 + c] + bhh[64 + jj * 16 + c];
    bxn[jj] = bih[128 + jj * 16 + c];
    bhn[jj] = bhh[128 + jj * 16 + c];
  }

  const int n0 = blockIdx.x * (16 * GRU_WAVES) + wave * 16;
  int nodeA = n0 + (lane & 15);
  if (nodeA > NN - 1) nodeA = NN - 1;               // clamp: keeps EXEC full for WMMA
  const float* xbase0 = x + (size_t)nodeA * ((SS + DD) * TT) + SS * TT;

  float hC[4][8];
#pragma unroll
  for (int jj = 0; jj < 4; ++jj)
#pragma unroll
    for (int rr = 0; rr < 8; ++rr) hC[jj][rr] = 0.f;

  for (int t = 0; t < TT; ++t) {
    FragU ax, ahl, ahh;
    const float* hrow = &s_h[wave][(lane & 15) * HH];
#pragma unroll
    for (int v = 0; v < 8; ++v) {
      const int k0 = kpair(v, halfc);
      ax.u[v]  = pk_bf16(xbase0[k0 * TT + t], xbase0[(k0 + 1) * TT + t]);
      ahl.u[v] = pk_bf16(hrow[k0], hrow[k0 + 1]);
      ahh.u[v] = pk_bf16(hrow[32 + k0], hrow[33 + k0]);
    }

#pragma unroll
    for (int jj = 0; jj < 4; ++jj) {
      // r gate (tile jj)
      v8f ar = {};
      ar = WMMA_BF16(ahl.v, load_frag_lds(&s_bfrag[(12 + jj * 2 + 0) * 256 + lane * 8]), ar);
      ar = WMMA_BF16(ahh.v, load_frag_lds(&s_bfrag[(12 + jj * 2 + 1) * 256 + lane * 8]), ar);
      ar = WMMA_BF16(ax.v,  load_frag_lds(&s_bfrag[jj * 256 + lane * 8]), ar);
      // z gate (tile 4+jj)
      v8f az_ = {};
      az_ = WMMA_BF16(ahl.v, load_frag_lds(&s_bfrag[(12 + (4 + jj) * 2 + 0) * 256 + lane * 8]), az_);
      az_ = WMMA_BF16(ahh.v, load_frag_lds(&s_bfrag[(12 + (4 + jj) * 2 + 1) * 256 + lane * 8]), az_);
      az_ = WMMA_BF16(ax.v,  load_frag_lds(&s_bfrag[(4 + jj) * 256 + lane * 8]), az_);
      // n gate: hn (tile 8+jj via Whh) and xn (via Wih) kept separate
      v8f ahn = {};
      ahn = WMMA_BF16(ahl.v, load_frag_lds(&s_bfrag[(12 + (8 + jj) * 2 + 0) * 256 + lane * 8]), ahn);
      ahn = WMMA_BF16(ahh.v, load_frag_lds(&s_bfrag[(12 + (8 + jj) * 2 + 1) * 256 + lane * 8]), ahn);
      v8f axn = {};
      axn = WMMA_BF16(ax.v,  load_frag_lds(&s_bfrag[(8 + jj) * 256 + lane * 8]), axn);

#pragma unroll
      for (int rr = 0; rr < 8; ++rr) {
        const float rv = fast_sig(ar[rr] + br[jj]);
        const float zv = fast_sig(az_[rr] + bz[jj]);
        const float nv = fast_tanh(axn[rr] + bxn[jj] + rv * (ahn[rr] + bhn[jj]));
        hC[jj][rr] = (1.f - zv) * nv + zv * hC[jj][rr];
      }
#pragma unroll
      for (int rr = 0; rr < 8; ++rr)
        s_h[wave][(rr + 8 * halfc) * HH + jj * 16 + c] = hC[jj][rr];
    }
    __syncthreads();
  }

#pragma unroll
  for (int jj = 0; jj < 4; ++jj)
#pragma unroll
    for (int rr = 0; rr < 8; ++rr) {
      const int node = n0 + rr + 8 * halfc;
      if (node < NN) gruh[(size_t)node * HH + jj * 16 + c] = hC[jj][rr];
    }
}

// --------------------------------------------------------------------------
// WMMA head kernel: gc2 (K=32) + LayerNorm + lin1 (K=128) + lin2 (K=64) + out.
// One wave = 16 nodes. Grid*32 == NN exactly -> EXEC always full.
// --------------------------------------------------------------------------
#define HD_WAVES 2
__global__ __launch_bounds__(32 * HD_WAVES)
void head_kernel(const float* __restrict__ gruh, const float* __restrict__ msg2,
                 const float* __restrict__ norm_dst,
                 const float* __restrict__ W2gc, const float* __restrict__ b2gc,
                 const float* __restrict__ lng, const float* __restrict__ lnb,
                 const float* __restrict__ W1, const float* __restrict__ b1,
                 const float* __restrict__ W2, const float* __restrict__ b2,
                 const float* __restrict__ oW, const float* __restrict__ ob,
                 float* __restrict__ out)
{
  // B tiles: 0..3 gcW2 (K=32, col jj) | 4..19 lin1 (4+jj*4+kk) | 20..27 lin2 (20+jj*2+kk)
  __shared__ unsigned s_bfrag[28 * 256];
  __shared__ float s_g[128], s_bb[128];
  __shared__ float s_z[HD_WAVES][16 * 128];
  __shared__ float s_a[HD_WAVES][16 * 64];
  __shared__ float s_red[HD_WAVES][64];
  __shared__ float s_logit[HD_WAVES][16];

  const int tid   = threadIdx.x;
  const int lane  = tid & 31;
  const int wave  = tid >> 5;
  const int c     = lane & 15;
  const int halfc = lane >> 4;

  for (int idx = tid; idx < 28 * 32; idx += blockDim.x) {
    const int fl = idx & 31;
    const int tile = idx >> 5;
    const int jcol = fl & 15;
    const int fh = fl >> 4;
    unsigned frag[8];
#pragma unroll
    for (int v = 0; v < 8; ++v) {
      const int k0 = kpair(v, fh);
      float f0, f1;
      if (tile < 4) {
        const int col = tile * 16 + jcol;
        f0 = W2gc[k0 * 64 + col];
        f1 = W2gc[(k0 + 1) * 64 + col];
      } else if (tile < 20) {
        const int tt2 = tile - 4;
        const int jj = tt2 >> 2, kk = tt2 & 3;
        const int col = jj * 16 + jcol;
        const int k = kk * 32 + k0;
        f0 = W1[k * 64 + col];
        f1 = W1[(k + 1) * 64 + col];
      } else {
        const int tt2 = tile - 20;
        const int jj = tt2 >> 1, kk = tt2 & 1;
        const int col = jj * 16 + jcol;
        const int k = kk * 32 + k0;
        f0 = W2[k * 64 + col];
        f1 = W2[(k + 1) * 64 + col];
      }
      frag[v] = pk_bf16(f0, f1);
    }
#pragma unroll
    for (int v = 0; v < 8; ++v) s_bfrag[tile * 256 + fl * 8 + v] = frag[v];
  }
  for (int i = tid; i < 128; i += blockDim.x) { s_g[i] = lng[i]; s_bb[i] = lnb[i]; }
  __syncthreads();

  float bgc[4], bl1[4], bl2[4], owr[4];
#pragma unroll
  for (int jj = 0; jj < 4; ++jj) {
    bgc[jj] = b2gc[jj * 16 + c];
    bl1[jj] = b1[jj * 16 + c];
    bl2[jj] = b2[jj * 16 + c];
    owr[jj] = oW[jj * 16 + c];
  }
  const float obv = ob[0];

  const int n0 = blockIdx.x * (16 * HD_WAVES) + wave * 16;   // 3125*32 == NN exactly

  // Stage gruh tile into s_z[:, 0:64] (coalesced float4)
  {
    const float4* gsrc = (const float4*)(gruh + (size_t)n0 * 64);
    for (int i = lane; i < 256; i += 32) {
      float4 v4 = gsrc[i];
      const int node = i >> 4;            // 16 float4 per node
      const int kq = (i & 15) * 4;
      float* d = &s_z[wave][node * 128 + kq];
      d[0] = v4.x; d[1] = v4.y; d[2] = v4.z; d[3] = v4.w;
    }
  }

  // gc2: A fragment = msg2 * norm_dst (K=32)
  const int nodeA = n0 + (lane & 15);
  const float nd = norm_dst[nodeA];
  const float* m2r = msg2 + (size_t)nodeA * 32;
  FragU am;
#pragma unroll
  for (int v = 0; v < 8; ++v) {
    const int k0 = kpair(v, halfc);
    am.u[v] = pk_bf16(m2r[k0] * nd, m2r[k0 + 1] * nd);
  }
#pragma unroll
  for (int jj = 0; jj < 4; ++jj) {
    v8f acc = {};
    acc = WMMA_BF16(am.v, load_frag_lds(&s_bfrag[jj * 256 + lane * 8]), acc);
#pragma unroll
    for (int rr = 0; rr < 8; ++rr) {
      const float hv = fmaxf(acc[rr] + bgc[jj], 0.f);
      s_z[wave][(rr + 8 * halfc) * 128 + 64 + jj * 16 + c] = hv;
    }
  }

  // LayerNorm stats: lane pair (l, l^16) splits the 128 features of node l&15
  {
    const float* zr = &s_z[wave][(lane & 15) * 128 + halfc * 64];
    float sum = 0.f, sq = 0.f;
#pragma unroll
    for (int k = 0; k < 64; ++k) { const float v = zr[k]; sum += v; sq += v * v; }
    s_red[wave][lane] = sum;
    s_red[wave][32 + lane] = sq;
  }
  const float tsum = s_red[wave][lane] + s_red[wave][lane ^ 16];
  const float tsq  = s_red[wave][32 + lane] + s_red[wave][32 + (lane ^ 16)];
  const float mu  = tsum * (1.f / 128.f);
  const float inv = rsqrtf(tsq * (1.f / 128.f) - mu * mu + 1e-5f);

  // z A-fragments (K=128 -> 4 frags), normalized + affine
  FragU az[4];
  {
    const float* zrow = &s_z[wave][(lane & 15) * 128];
#pragma unroll
    for (int f = 0; f < 4; ++f)
#pragma unroll
      for (int v = 0; v < 8; ++v) {
        const int k = f * 32 + kpair(v, halfc);
        const float z0 = (zrow[k] - mu) * inv * s_g[k] + s_bb[k];
        const float z1 = (zrow[k + 1] - mu) * inv * s_g[k + 1] + s_bb[k + 1];
        az[f].u[v] = pk_bf16(z0, z1);
      }
  }

  // lin1: 4 col tiles x 4 K-frags
#pragma unroll
  for (int jj = 0; jj < 4; ++jj) {
    v8f acc = {};
#pragma unroll
    for (int kk = 0; kk < 4; ++kk)
      acc = WMMA_BF16(az[kk].v, load_frag_lds(&s_bfrag[(4 + jj * 4 + kk) * 256 + lane * 8]), acc);
#pragma unroll
    for (int rr = 0; rr < 8; ++rr)
      s_a[wave][(rr + 8 * halfc) * 64 + jj * 16 + c] = fmaxf(acc[rr] + bl1[jj], 0.f);
  }

  // a A-fragments (K=64 -> 2 frags)
  FragU aa[2];
  {
    const float* arow = &s_a[wave][(lane & 15) * 64];
#pragma unroll
    for (int f = 0; f < 2; ++f)
#pragma unroll
      for (int v = 0; v < 8; ++v) {
        const int k = f * 32 + kpair(v, halfc);
        aa[f].u[v] = pk_bf16(arow[k], arow[k + 1]);
      }
  }

  if (lane < 16) s_logit[wave][lane] = 0.f;

  // lin2: 4 col tiles x 2 K-frags; hidden -> d_out, logit partials via ds_add_f32
  float lp[8];
#pragma unroll
  for (int rr = 0; rr < 8; ++rr) lp[rr] = 0.f;
#pragma unroll
  for (int jj = 0; jj < 4; ++jj) {
    v8f acc = {};
#pragma unroll
    for (int kk = 0; kk < 2; ++kk)
      acc = WMMA_BF16(aa[kk].v, load_frag_lds(&s_bfrag[(20 + jj * 2 + kk) * 256 + lane * 8]), acc);
#pragma unroll
    for (int rr = 0; rr < 8; ++rr) {
      const float hv = fmaxf(acc[rr] + bl2[jj], 0.f);
      const int node = n0 + rr + 8 * halfc;
      out[(size_t)2 * NN + (size_t)node * 64 + jj * 16 + c] = hv;
      lp[rr] += hv * owr[jj];
    }
  }
#pragma unroll
  for (int rr = 0; rr < 8; ++rr)
    atomicAdd(&s_logit[wave][rr + 8 * halfc], lp[rr]);   // 16 column-lanes reduce per node

  if (lane < 16) {
    const int n = n0 + lane;
    const float lg = s_logit[wave][lane] + obv;
    out[n] = lg;
    out[NN + n] = lg;
  }
}

// --------------------------------------------------------------------------
// Graph-conv support kernels (L2-resident scatter, small VALU matmul)
// --------------------------------------------------------------------------
__global__ void zero_kernel(float* __restrict__ p, int n) {
  int i = blockIdx.x * blockDim.x + threadIdx.x;
  if (i < n) p[i] = 0.f;
}

__global__ void degree_kernel(const int* __restrict__ g,
                              float* __restrict__ deg_out, float* __restrict__ deg_in) {
  int e = blockIdx.x * blockDim.x + threadIdx.x;
  if (e >= EE) return;
  atomicAdd(&deg_out[g[e]], 1.f);
  atomicAdd(&deg_in[g[EE + e]], 1.f);
}

__global__ void norm_h0_kernel(const float* __restrict__ x,
                               float* __restrict__ deg_out, float* __restrict__ deg_in,
                               float* __restrict__ h0s) {
  int n = blockIdx.x * blockDim.x + threadIdx.x;
  if (n >= NN) return;
  float ns = rsqrtf(fmaxf(deg_out[n], 1.f));
  float ndv = rsqrtf(fmaxf(deg_in[n], 1.f));
  deg_out[n] = ns;  // -> norm_src
  deg_in[n]  = ndv; // -> norm_dst
  const float* xb = x + (size_t)n * ((SS + DD) * TT);
#pragma unroll
  for (int s = 0; s < SS; ++s) h0s[(size_t)n * SS + s] = xb[s * TT] * ns;
}

__global__ void scatter1_kernel(const int* __restrict__ g, const float* __restrict__ h0s,
                                float* __restrict__ msg1) {
  int i = blockIdx.x * blockDim.x + threadIdx.x;
  if (i >= EE * SS) return;
  int e = i >> 4, s = i & 15;
  atomicAdd(&msg1[(size_t)g[EE + e] * SS + s], h0s[(size_t)g[e] * SS + s]);
}

__global__ __launch_bounds__(256)
void gc1_kernel(const float* __restrict__ msg1, const float* __restrict__ norm_dst,
                const float* __restrict__ norm_src,
                const float* __restrict__ W1, const float* __restrict__ b1,
                float* __restrict__ h1s) {
  __shared__ float sW[SS * 32];
  __shared__ float sb[32];
  for (int i = threadIdx.x; i < SS * 32; i += blockDim.x) sW[i] = W1[i];
  if (threadIdx.x < 32) sb[threadIdx.x] = b1[threadIdx.x];
  __syncthreads();
  int n = blockIdx.x * blockDim.x + threadIdx.x;
  if (n >= NN) return;
  float m[SS];
  float ndv = norm_dst[n];
#pragma unroll
  for (int s = 0; s < SS; ++s) m[s] = msg1[(size_t)n * SS + s] * ndv;
  float ns = norm_src[n];
#pragma unroll
  for (int j = 0; j < 32; ++j) {
    float acc = sb[j];
#pragma unroll
    for (int s = 0; s < SS; ++s) acc += m[s] * sW[s * 32 + j];
    h1s[(size_t)n * 32 + j] = fmaxf(acc, 0.f) * ns;   // pre-scaled for layer 2
  }
}

__global__ void scatter2_kernel(const int* __restrict__ g, const float* __restrict__ h1s,
                                float* __restrict__ msg2) {
  int i = blockIdx.x * blockDim.x + threadIdx.x;
  if (i >= EE * 32) return;
  int e = i >> 5, j = i & 31;
  atomicAdd(&msg2[(size_t)g[EE + e] * 32 + j], h1s[(size_t)g[e] * 32 + j]);
}

// --------------------------------------------------------------------------
extern "C" void kernel_launch(void* const* d_in, const int* in_sizes, int n_in,
                              void* d_out, int out_size, void* d_ws, size_t ws_size,
                              hipStream_t stream) {
  (void)in_sizes; (void)n_in; (void)out_size; (void)ws_size;
  const float* x    = (const float*)d_in[0];
  const int*   grp  = (const int*)d_in[1];
  const float* Wih  = (const float*)d_in[2];
  const float* Whh  = (const float*)d_in[3];
  const float* bih  = (const float*)d_in[4];
  const float* bhh  = (const float*)d_in[5];
  const float* W1   = (const float*)d_in[6];
  const float* b1   = (const float*)d_in[7];
  const float* W2   = (const float*)d_in[8];
  const float* b2   = (const float*)d_in[9];
  const float* lng  = (const float*)d_in[10];
  const float* lnb  = (const float*)d_in[11];
  const float* l1W  = (const float*)d_in[12];
  const float* l1b  = (const float*)d_in[13];
  const float* l2W  = (const float*)d_in[14];
  const float* l2b  = (const float*)d_in[15];
  const float* oW   = (const float*)d_in[16];
  const float* ob   = (const float*)d_in[17];
  float* out = (float*)d_out;
  float* ws  = (float*)d_ws;

  // Workspace layout (floats); atomically-accumulated regions first (zeroed).
  float* deg_out = ws;                     // N   (-> norm_src)
  float* deg_in  = ws + (size_t)NN;        // N   (-> norm_dst)
  float* msg1    = ws + (size_t)2 * NN;    // 16N
  float* msg2    = ws + (size_t)18 * NN;   // 32N
  float* h1s     = ws + (size_t)50 * NN;   // 32N
  float* gruh    = ws + (size_t)82 * NN;   // 64N
  float* h0s     = ws + (size_t)146 * NN;  // 16N  (total 162N floats ~ 65 MB)

  const int B = 256;
  const int zn = 50 * NN;
  zero_kernel<<<(zn + B - 1) / B, B, 0, stream>>>(ws, zn);
  degree_kernel<<<(EE + B - 1) / B, B, 0, stream>>>(grp, deg_out, deg_in);
  norm_h0_kernel<<<(NN + B - 1) / B, B, 0, stream>>>(x, deg_out, deg_in, h0s);
  scatter1_kernel<<<(EE * SS + B - 1) / B, B, 0, stream>>>(grp, h0s, msg1);
  gc1_kernel<<<(NN + B - 1) / B, B, 0, stream>>>(msg1, deg_in, deg_out, W1, b1, h1s);
  scatter2_kernel<<<(EE * 32 + B - 1) / B, B, 0, stream>>>(grp, h1s, msg2);
  gru_kernel<<<(NN + 16 * GRU_WAVES - 1) / (16 * GRU_WAVES), 32 * GRU_WAVES, 0, stream>>>(
      x, Wih, Whh, bih, bhh, gruh);
  head_kernel<<<NN / (16 * HD_WAVES), 32 * HD_WAVES, 0, stream>>>(
      gruh, msg2, deg_in, W2, b2, lng, lnb, l1W, l1b, l2W, l2b, oW, ob, out);
}
